// SelfAttentionBlock_12549894439654
// MI455X (gfx1250) — matrix operand order repaired and checked by reading, not compiled, not run on hardware
//
#include <hip/hip_runtime.h>
#include <hip/hip_bf16.h>
#include <type_traits>

// ---------------------------------------------------------------------------
// CDNA5 (gfx1250) fused self-attention block, bf16 WMMA everywhere.
//   cbr(qf,W1)->tmp16; cbr(tmp16,W2)->q16; cbr(kf,W1)->tmp16; cbr(tmp16,W2)->k16;
//   cbr(kf,Wv)->v16;   flash-attn(q16,k16,v16)->ctx16;  cbr(ctx16,Wo)->out(f32)
// Intermediates live in workspace as bf16 (halves HBM traffic, kills cvt VALU).
// LDS operand layout: [row][kpair] dwords, padded row stride (multiple of 4),
// so every WMMA fragment is two aligned ds_load_b128 into contiguous VGPRs.
// V-tile staging uses global_load_async_to_lds_b128 (+ s_wait_asynccnt).
// ---------------------------------------------------------------------------

typedef __attribute__((ext_vector_type(16))) __bf16 v16bf;
typedef __attribute__((ext_vector_type(2)))  __bf16 v2bf;
typedef __attribute__((ext_vector_type(8)))  float  v8f;

union FragBF { unsigned u[8]; v16bf v; };
union FragF  { float    f[8]; v8f   v; };

__device__ __forceinline__ unsigned short f2bf(float x) {
    return __builtin_bit_cast(unsigned short, (__bf16)x);   // native v_cvt, RNE
}
__device__ __forceinline__ unsigned packbf(float lo, float hi) {
    v2bf p; p[0] = (__bf16)lo; p[1] = (__bf16)hi;
    return __builtin_bit_cast(unsigned, p);
}
__device__ __forceinline__ void ld_frag(FragBF& d, const unsigned* base, int i0, int i1) {
    *reinterpret_cast<uint4*>(&d.u[0]) = *reinterpret_cast<const uint4*>(base + i0);
    *reinterpret_cast<uint4*>(&d.u[4]) = *reinterpret_cast<const uint4*>(base + i1);
}

#define BN_SCALE 0.9999950000374997f                 // 1/sqrt(1+1e-5)

// ---------------------------------------------------------------------------
// Fused 1x1conv + BN(eval) + ReLU:  Y[m,n] = relu(scale[m]*sum_k W[m,k]X[k,n] + b[m])
// TIN: float (raw feats) or unsigned short (bf16 intermediates).
// OUT_BF16: bf16 intermediate output vs fp32 final output.
// Block 256 thr (8 waves); tile 64x128, BK=32 (one WMMA K step).
// ---------------------------------------------------------------------------
#define GM 64
#define GN 128
#define GK 32
#define SAW 20                                       // sA row stride (dwords)
#define SBW 20                                       // sB row stride (dwords)

template <typename TIN, bool OUT_BF16>
__global__ __launch_bounds__(256)
void cbr_gemm_bf16(const float* __restrict__ W, const TIN* __restrict__ X,
                   const float* __restrict__ g, const float* __restrict__ bia,
                   void* __restrict__ Yv, int M, int N, int K) {
    __shared__ unsigned sA32[GM * SAW];              // [m][kpair], bf16 pairs
    __shared__ unsigned sB32[GN * SBW];              // [n][kpair], bf16 pairs

    const int tid  = threadIdx.x;
    const int wave = tid >> 5, lane = tid & 31;
    const int lan  = lane & 15, half = lane >> 4;
    const int msub = wave & 3,  nhlf = wave >> 2;
    const int m0 = blockIdx.y * GM, n0 = blockIdx.x * GN;
    const TIN* Xb = X + (size_t)blockIdx.z * K * N;

    FragF acc[4];
#pragma unroll
    for (int f = 0; f < 4; ++f)
#pragma unroll
        for (int v = 0; v < 8; ++v) acc[f].f[v] = 0.0f;

    for (int k0 = 0; k0 < K; k0 += GK) {
        // ---- stage A 64x32: float4 loads, pack k-pairs (W is always fp32) ----
#pragma unroll
        for (int p = 0; p < 2; ++p) {
            const int i = tid + p * 256;             // 512 float4s
            const int m = i >> 3, c = i & 7;
            const float4 w4 = *reinterpret_cast<const float4*>(
                W + (size_t)(m0 + m) * K + k0 + 4 * c);
            sA32[m * SAW + 2 * c]     = packbf(w4.x, w4.y);
            sA32[m * SAW + 2 * c + 1] = packbf(w4.z, w4.w);
        }
        // ---- stage B 32x128: rows (k,k+1), pack pairs over k ----
#pragma unroll
        for (int p = 0; p < 2; ++p) {
            const int i = tid + p * 256;             // 512 pair-groups
            const int r = i >> 5, c = i & 31;        // r = k/2, c = n/4
            if constexpr (std::is_same<TIN, float>::value) {
                const float4 x0 = *reinterpret_cast<const float4*>(
                    Xb + (size_t)(k0 + 2 * r) * N + n0 + 4 * c);
                const float4 x1 = *reinterpret_cast<const float4*>(
                    Xb + (size_t)(k0 + 2 * r + 1) * N + n0 + 4 * c);
                sB32[(4 * c + 0) * SBW + r] = packbf(x0.x, x1.x);
                sB32[(4 * c + 1) * SBW + r] = packbf(x0.y, x1.y);
                sB32[(4 * c + 2) * SBW + r] = packbf(x0.z, x1.z);
                sB32[(4 * c + 3) * SBW + r] = packbf(x0.w, x1.w);
            } else {                                 // bf16 input: pure repack
                const ushort4 x0 = *reinterpret_cast<const ushort4*>(
                    Xb + (size_t)(k0 + 2 * r) * N + n0 + 4 * c);
                const ushort4 x1 = *reinterpret_cast<const ushort4*>(
                    Xb + (size_t)(k0 + 2 * r + 1) * N + n0 + 4 * c);
                sB32[(4 * c + 0) * SBW + r] = (unsigned)x0.x | ((unsigned)x1.x << 16);
                sB32[(4 * c + 1) * SBW + r] = (unsigned)x0.y | ((unsigned)x1.y << 16);
                sB32[(4 * c + 2) * SBW + r] = (unsigned)x0.z | ((unsigned)x1.z << 16);
                sB32[(4 * c + 3) * SBW + r] = (unsigned)x0.w | ((unsigned)x1.w << 16);
            }
        }
        if (k0 + GK < K)                             // global_prefetch_b8
            __builtin_prefetch(Xb + (size_t)(k0 + GK + (tid >> 3)) * N + n0 + (tid & 7) * 16, 0, 1);
        __syncthreads();

        // ---- fragments: two aligned b128 each ----
        FragBF a;
        const int mrow = msub * 16 + lan;
        ld_frag(a, sA32, mrow * SAW + half * 4, mrow * SAW + 8 + half * 4);
#pragma unroll
        for (int f = 0; f < 4; ++f) {
            FragBF b;
            const int ncol = nhlf * 64 + f * 16 + lan;
            ld_frag(b, sB32, ncol * SBW + half * 4, ncol * SBW + 8 + half * 4);
            acc[f].v = __builtin_amdgcn_wmma_f32_16x16x32_bf16(
                false, a.v, false, b.v, (short)0, acc[f].v, false, false);
        }
        __syncthreads();
    }

    // ---- epilogue: BN + ReLU; D rows m = v+8*half, cols n = lan ----
    float sc[8], bs[8];
#pragma unroll
    for (int v = 0; v < 8; ++v) {
        const int mrow = m0 + msub * 16 + v + 8 * half;
        sc[v] = g[mrow] * BN_SCALE;
        bs[v] = bia[mrow];
    }
#pragma unroll
    for (int f = 0; f < 4; ++f) {
        const int ncol = n0 + nhlf * 64 + f * 16 + lan;
#pragma unroll
        for (int v = 0; v < 8; ++v) {
            const int mrow = m0 + msub * 16 + v + 8 * half;
            const float y = fmaxf(acc[f].f[v] * sc[v] + bs[v], 0.0f);
            if constexpr (OUT_BF16)
                ((unsigned short*)Yv)[(size_t)blockIdx.z * M * N + (size_t)mrow * N + ncol] = f2bf(y);
            else
                ((float*)Yv)[(size_t)blockIdx.z * M * N + (size_t)mrow * N + ncol] = y;
        }
    }
}

// ---------------------------------------------------------------------------
// Fused flash-attention: ctx[t,n] = sum_m softmax_m(q^T k /16)[n,m] v[t,m]
// q,k,v,ctx: [Bz, T=256, N] bf16. Block 256 thr, 64 query rows, 32-key tiles.
// Wave w: n-subtile = w&3; S col-tile / O t-half = w>>2.
// ---------------------------------------------------------------------------
#define AROWS 64
#define AMT   32
#define AT    256
#define SKW   132                                    // sK row stride (dwords)
#define SVW   20                                     // sV row stride (dwords)
#define SPW   12                                     // sP row stride (dwords)

__global__ __launch_bounds__(256)
void attn_fused(const unsigned short* __restrict__ q, const unsigned short* __restrict__ k,
                const unsigned short* __restrict__ v, unsigned short* __restrict__ ctx, int N) {
    __shared__ unsigned sK32[AMT * SKW];             // [m][tpair]
    __shared__ unsigned sV32[AT * SVW];              // [t][mpair]
    __shared__ float    sS[AROWS * AMT];
    __shared__ unsigned sP32[AROWS * SPW];           // [n][mpair]
    __shared__ float sM[AROWS], sL[AROWS], sAl[AROWS], sRed[AROWS * 4];

    const int tid  = threadIdx.x;
    const int wave = tid >> 5, lane = tid & 31;
    const int lan  = lane & 15, half = lane >> 4;
    const int ntile = wave & 3, wgrp = wave >> 2;
    const int n0 = blockIdx.x * AROWS;
    const size_t bo = (size_t)blockIdx.y * AT * N;
    const unsigned short* qb = q + bo;
    const unsigned short* kb = k + bo;
    const unsigned short* vb = v + bo;
    unsigned short*       cb = ctx + bo;

    // ---- preload q A-fragments; base absorbs runtime 'half' so the t-offsets
    //      become compile-time immediates in the global loads ----
    unsigned aq[8][8];
    {
        const unsigned short* qp0 = qb + (n0 + ntile * 16 + lan) + (size_t)half * 8 * N;
#pragma unroll
        for (int s = 0; s < 8; ++s)
#pragma unroll
            for (int vv = 0; vv < 8; ++vv) {
                const int tofs = s * 32 + (vv >> 2) * 16 + (vv & 3) * 2;
                aq[s][vv] = (unsigned)qp0[(size_t)tofs * N] |
                            ((unsigned)qp0[(size_t)(tofs + 1) * N] << 16);
            }
    }

    FragF o[8];
#pragma unroll
    for (int f = 0; f < 8; ++f)
#pragma unroll
        for (int vv = 0; vv < 8; ++vv) o[f].f[vv] = 0.0f;
    if (tid < AROWS) { sM[tid] = -1e30f; sL[tid] = 0.0f; }
    __syncthreads();

    for (int m0 = 0; m0 < N; m0 += AMT) {
        // ---- stage K (pack pairs over t): ushort2 over m, rows t,t+1 ----
        {
            const int mg = tid & 15, rb = tid >> 4;  // mg: m/2, rb: base t-pair
#pragma unroll
            for (int p = 0; p < 8; ++p) {
                const int r = rb + p * 16;           // t-pair 0..127
                const ushort2 a2 = *reinterpret_cast<const ushort2*>(
                    kb + (size_t)(2 * r) * N + m0 + 2 * mg);
                const ushort2 b2 = *reinterpret_cast<const ushort2*>(
                    kb + (size_t)(2 * r + 1) * N + m0 + 2 * mg);
                sK32[(2 * mg + 0) * SKW + r] = (unsigned)a2.x | ((unsigned)b2.x << 16);
                sK32[(2 * mg + 1) * SKW + r] = (unsigned)a2.y | ((unsigned)b2.y << 16);
            }
        }
        // ---- stage V: layout-preserving bf16 copy -> async LDS DMA (b128) ----
#pragma unroll
        for (int p = 0; p < 4; ++p) {
            const int i = tid + p * 256;             // 1024 16-byte chunks
            const int t = i >> 2, cq = i & 3;        // row t, quarter (8 m's)
            const unsigned lds_dst = (unsigned)(size_t)&sV32[t * SVW + 4 * cq];
            const unsigned short* gsrc = vb + (size_t)t * N + m0 + 8 * cq;
            asm volatile("global_load_async_to_lds_b128 %0, %1, off"
                         :: "v"(lds_dst), "v"(gsrc) : "memory");
        }
        asm volatile("s_wait_asynccnt 0x0" ::: "memory");
        __syncthreads();

        // ---- S = (q^T k)/16 : one 16x16 frag per wave, K=256 (8 steps) ----
        FragF sac;
#pragma unroll
        for (int vv = 0; vv < 8; ++vv) sac.f[vv] = 0.0f;
        const int mcol = wgrp * 16 + lan;
#pragma unroll
        for (int s = 0; s < 8; ++s) {
            FragBF a, b;
#pragma unroll
            for (int vv = 0; vv < 8; ++vv) a.u[vv] = aq[s][vv];
            const int bi = mcol * SKW + s * 16 + half * 4;
            ld_frag(b, sK32, bi, bi + 8);
            sac.v = __builtin_amdgcn_wmma_f32_16x16x32_bf16(
                false, a.v, false, b.v, (short)0, sac.v, false, false);
        }
#pragma unroll
        for (int vv = 0; vv < 8; ++vv)
            sS[(ntile * 16 + vv + 8 * half) * AMT + wgrp * 16 + lan] = sac.f[vv] * 0.0625f;
        __syncthreads();

        // ---- online softmax (4 threads per row, 8 cols each) ----
        {
            const int row = tid >> 2, qq = tid & 3;
            float mx = -1e30f;
#pragma unroll
            for (int c = 0; c < 8; ++c) mx = fmaxf(mx, sS[row * AMT + qq * 8 + c]);
            sRed[row * 4 + qq] = mx;
        }
        __syncthreads();
        if (tid < AROWS) {
            const float tm = fmaxf(fmaxf(sRed[tid * 4], sRed[tid * 4 + 1]),
                                   fmaxf(sRed[tid * 4 + 2], sRed[tid * 4 + 3]));
            const float mo = sM[tid], mn = fmaxf(mo, tm);
            sAl[tid] = __expf(mo - mn);
            sM[tid]  = mn;
        }
        __syncthreads();
        {
            const int row = tid >> 2, qq = tid & 3;
            const float mn = sM[row];
            unsigned short* sP16 = reinterpret_cast<unsigned short*>(sP32);
            float ssum = 0.0f;
#pragma unroll
            for (int c = 0; c < 8; ++c) {
                const float p = __expf(sS[row * AMT + qq * 8 + c] - mn);
                ssum += p;
                sP16[row * (2 * SPW) + qq * 8 + c] = f2bf(p);
            }
            sRed[row * 4 + qq] = ssum;
        }
        __syncthreads();
        if (tid < AROWS)
            sL[tid] = sAl[tid] * sL[tid] +
                      sRed[tid * 4] + sRed[tid * 4 + 1] + sRed[tid * 4 + 2] + sRed[tid * 4 + 3];

        // ---- O = alpha*O + P.V : 8 t-frags per wave, K=32 (one step) ----
        float av[8];
#pragma unroll
        for (int vv = 0; vv < 8; ++vv) av[vv] = sAl[ntile * 16 + vv + 8 * half];
        FragBF pa;
        {
            const int nloc = ntile * 16 + lan;
            ld_frag(pa, sP32, nloc * SPW + half * 4, nloc * SPW + 8 + half * 4);
        }
#pragma unroll
        for (int f = 0; f < 8; ++f) {
#pragma unroll
            for (int vv = 0; vv < 8; ++vv) o[f].f[vv] *= av[vv];
            FragBF bv;
            const int t = wgrp * 128 + f * 16 + lan;
            ld_frag(bv, sV32, t * SVW + half * 4, t * SVW + 8 + half * 4);
            o[f].v = __builtin_amdgcn_wmma_f32_16x16x32_bf16(
                false, pa.v, false, bv.v, (short)0, o[f].v, false, false);
        }
        __syncthreads();
    }

    // ---- normalize by l and store ctx[t][n] (bf16) ----
    float lv[8];
#pragma unroll
    for (int vv = 0; vv < 8; ++vv) lv[vv] = 1.0f / sL[ntile * 16 + vv + 8 * half];
#pragma unroll
    for (int f = 0; f < 8; ++f) {
        const int t = wgrp * 128 + f * 16 + lan;
#pragma unroll
        for (int vv = 0; vv < 8; ++vv) {
            const int n = n0 + ntile * 16 + vv + 8 * half;
            cb[(size_t)t * N + n] = f2bf(o[f].f[vv] * lv[vv]);
        }
    }
}

// ---------------------------------------------------------------------------
extern "C" void kernel_launch(void* const* d_in, const int* in_sizes, int n_in,
                              void* d_out, int out_size, void* d_ws, size_t ws_size,
                              hipStream_t stream) {
    const int B = 4, C = 512, T = 256, O = 512, N = 64 * 64;

    const float* qf    = (const float*)d_in[0];
    const float* kf    = (const float*)d_in[1];
    const float* qp_w1 = (const float*)d_in[2];
    const float* qp_g1 = (const float*)d_in[3];
    const float* qp_b1 = (const float*)d_in[4];
    const float* qp_w2 = (const float*)d_in[5];
    const float* qp_g2 = (const float*)d_in[6];
    const float* qp_b2 = (const float*)d_in[7];
    const float* kp_w1 = (const float*)d_in[8];
    const float* kp_g1 = (const float*)d_in[9];
    const float* kp_b1 = (const float*)d_in[10];
    const float* kp_w2 = (const float*)d_in[11];
    const float* kp_g2 = (const float*)d_in[12];
    const float* kp_b2 = (const float*)d_in[13];
    const float* vp_w  = (const float*)d_in[14];
    const float* vp_g  = (const float*)d_in[15];
    const float* vp_b  = (const float*)d_in[16];
    const float* op_w  = (const float*)d_in[17];
    const float* op_g  = (const float*)d_in[18];
    const float* op_b  = (const float*)d_in[19];

    const size_t bufN = (size_t)B * T * N;           // elements per bf16 buffer
    unsigned short* bq = (unsigned short*)d_ws;
    unsigned short* bk = bq + bufN;
    unsigned short* bv = bk + bufN;
    unsigned short* bt = bv + bufN;                  // stage-1 tmp / ctx (reused)

    const dim3 blk(256);
    const dim3 gP(N / GN, T / GM, B);                // projections: M=256
    const dim3 gO(N / GN, O / GM, B);                // out_project: M=512
    const dim3 gA(N / AROWS, B);

    cbr_gemm_bf16<float, true><<<gP, blk, 0, stream>>>(qp_w1, qf, qp_g1, qp_b1, bt, T, N, C);
    cbr_gemm_bf16<unsigned short, true><<<gP, blk, 0, stream>>>(qp_w2, bt, qp_g2, qp_b2, bq, T, N, T);
    cbr_gemm_bf16<float, true><<<gP, blk, 0, stream>>>(kp_w1, kf, kp_g1, kp_b1, bt, T, N, C);
    cbr_gemm_bf16<unsigned short, true><<<gP, blk, 0, stream>>>(kp_w2, bt, kp_g2, kp_b2, bk, T, N, T);
    cbr_gemm_bf16<float, true><<<gP, blk, 0, stream>>>(vp_w, kf, vp_g, vp_b, bv, T, N, C);
    attn_fused<<<gA, blk, 0, stream>>>(bq, bk, bv, bt, N);
    cbr_gemm_bf16<unsigned short, false><<<gO, blk, 0, stream>>>(op_w, bt, op_g, op_b, d_out, O, N, T);
}
